// DeepPoseGCN_86431921865001
// MI455X (gfx1250) — matrix operand order repaired and checked by reading, not compiled, not run on hardware
//
#include <hip/hip_runtime.h>

typedef __attribute__((ext_vector_type(16))) _Float16 v16h;
typedef __attribute__((ext_vector_type(8)))  float    v8f;

#define NNODES  100000            // divisible by 16 -> no row guards in WMMA GEMM
#define NEDGES  3200000
#define NGRAPHS 1024
#define BN_EPS  1e-5f
#define NC      (NNODES * 64)
#define NSLABS  (NNODES / 16)

// ---------------- degree / normalization ----------------
__global__ void k_init_deg(float* __restrict__ deg) {
    int i = blockIdx.x * blockDim.x + threadIdx.x;
    if (i < NNODES) deg[i] = 1.0f;                 // self-loop
}

__global__ void k_count_deg(const int* __restrict__ dst, float* __restrict__ deg) {
    int i = blockIdx.x * blockDim.x + threadIdx.x;
    if (i < NEDGES) atomicAdd(&deg[dst[i]], 1.0f);
}

__global__ void k_inv_sqrt(const float* __restrict__ deg, float* __restrict__ isq) {
    int i = blockIdx.x * blockDim.x + threadIdx.x;
    if (i < NNODES) isq[i] = rsqrtf(deg[i]);       // deg >= 1 always (self-loop)
}

// ---------------- layer 1: x[N,3] @ W1[3,64], fused self-loop+bias ----------------
__global__ void k_lin1(const float* __restrict__ x, const float* __restrict__ W1,
                       const float* __restrict__ isq, const float* __restrict__ b,
                       float* __restrict__ ht, float* __restrict__ agg) {
    int idx = blockIdx.x * blockDim.x + threadIdx.x;
    if (idx >= NC) return;
    int i = idx >> 6, c = idx & 63;
    float v = x[i*3+0]*W1[c] + x[i*3+1]*W1[64+c] + x[i*3+2]*W1[128+c];
    float s = isq[i];
    ht[idx]  = v;
    agg[idx] = v * s * s + b[c];
}

// ---------------- weight -> WMMA B-fragment (f16) prep ----------------
// Fragment layout: [tileN(4)][kchunk(2)][lane(32)][j(16)] halfs.
// lane l holds column n = tile*16 + (l&15); k = chunk*32 + (l>>4)*16 + j.
__global__ void k_prep_wfrag(const float* __restrict__ W2, const float* __restrict__ W3,
                             const float* __restrict__ W4, _Float16* __restrict__ wf) {
    int gid = blockIdx.x * blockDim.x + threadIdx.x;
    if (gid >= 3 * 4096) return;
    const float* W = (gid < 4096) ? W2 : (gid < 8192 ? W3 : W4);
    int r    = gid & 4095;
    int tile = r >> 10;
    int chnk = (r >> 9) & 1;
    int lane = (r >> 4) & 31;
    int j    = r & 15;
    int ncol = tile * 16 + (lane & 15);
    int k    = chnk * 32 + (lane >> 4) * 16 + j;
    wf[gid]  = (_Float16)W[k * 64 + ncol];
}

// ---------------- WMMA GEMM: h[N,64] @ W[64,64] -> ht, agg = ht*isq^2 + bias ------
// One wave per 16-row slab; 4 accumulator tiles (full 64 cols); K in 2 chunks of 32.
// N % 16 == 0 -> no per-row guards, no clamps; only whole-wave tail exit.
__global__ void k_gemm64_wmma(const float* __restrict__ h, const _Float16* __restrict__ wf,
                              const float* __restrict__ isq, const float* __restrict__ b,
                              float* __restrict__ ht, float* __restrict__ agg) {
    int wave = blockIdx.x * (blockDim.x >> 5) + (threadIdx.x >> 5);
    if (wave >= NSLABS) return;
    int lane = threadIdx.x & 31;
    int hi   = lane >> 4;
    int rowBase = wave << 4;
    int aRow = rowBase + (lane & 15);

    v8f acc0 = {}, acc1 = {}, acc2 = {}, acc3 = {};
#pragma unroll
    for (int c = 0; c < 2; ++c) {
        // A fragment: lane l holds row (l&15); k = c*32 + hi*16 + j
        const float* rp = h + (size_t)aRow * 64 + c * 32 + hi * 16;
        v16h a;
#pragma unroll
        for (int j = 0; j < 8; ++j) {
            a[j]     = (_Float16)rp[j];
            a[j + 8] = (_Float16)rp[8 + j];
        }
        v16h b0 = *(const v16h*)(wf + ((0 * 2 + c) * 32 + lane) * 16);
        v16h b1 = *(const v16h*)(wf + ((1 * 2 + c) * 32 + lane) * 16);
        v16h b2 = *(const v16h*)(wf + ((2 * 2 + c) * 32 + lane) * 16);
        v16h b3 = *(const v16h*)(wf + ((3 * 2 + c) * 32 + lane) * 16);
        acc0 = __builtin_amdgcn_wmma_f32_16x16x32_f16(false, a, false, b0, (short)0, acc0, false, false);
        acc1 = __builtin_amdgcn_wmma_f32_16x16x32_f16(false, a, false, b1, (short)0, acc1, false, false);
        acc2 = __builtin_amdgcn_wmma_f32_16x16x32_f16(false, a, false, b2, (short)0, acc2, false, false);
        acc3 = __builtin_amdgcn_wmma_f32_16x16x32_f16(false, a, false, b3, (short)0, acc3, false, false);
    }
    // D layout: lane 0-15 -> N=lane, VGPR r -> M=r (+8 for lanes 16-31)
    int col = lane & 15;
    float bi0 = b[col], bi1 = b[16 + col], bi2 = b[32 + col], bi3 = b[48 + col];
#pragma unroll
    for (int r = 0; r < 8; ++r) {
        int m = rowBase + hi * 8 + r;
        float s  = isq[m];
        float ss = s * s;
        float* oh = ht  + (size_t)m * 64 + col;
        float* oa = agg + (size_t)m * 64 + col;
        oh[0]  = acc0[r];            oa[0]  = acc0[r] * ss + bi0;
        oh[16] = acc1[r];            oa[16] = acc1[r] * ss + bi1;
        oh[32] = acc2[r];            oa[32] = acc2[r] * ss + bi2;
        oh[48] = acc3[r];            oa[48] = acc3[r] * ss + bi3;
    }
}

// ---------------- edge gather-scale-scatter (16 lanes / edge, float4) ----------------
__global__ void k_scatter(const float* __restrict__ ht, const int* __restrict__ src,
                          const int* __restrict__ dst, const float* __restrict__ isq,
                          float* __restrict__ agg) {
    int t  = blockIdx.x * blockDim.x + threadIdx.x;
    int ed = t >> 4;
    if (ed >= NEDGES) return;
    int c4 = (t & 15) << 2;
    int s = src[ed], d = dst[ed];
    float coef = isq[s] * isq[d];
    const float4 v = *(const float4*)(ht + (size_t)s * 64 + c4);
    float* o = agg + (size_t)d * 64 + c4;
    atomicAdd(o + 0, v.x * coef);
    atomicAdd(o + 1, v.y * coef);
    atomicAdd(o + 2, v.z * coef);
    atomicAdd(o + 3, v.w * coef);
}

// ---------------- BN(eval) + ReLU + optional residual ----------------
__global__ void k_bnrelu(const float* __restrict__ agg, const float* __restrict__ g,
                         const float* __restrict__ be, const float* __restrict__ m,
                         const float* __restrict__ v, const float* __restrict__ res,
                         float* __restrict__ out) {
    int idx = blockIdx.x * blockDim.x + threadIdx.x;
    if (idx >= NC) return;
    int c = idx & 63;
    float t = (agg[idx] - m[c]) * rsqrtf(v[c] + BN_EPS) * g[c] + be[c];
    t = fmaxf(t, 0.0f);
    if (res) t += res[idx];
    out[idx] = t;
}

// ---------------- pooling ----------------
__global__ void k_pool_zero(float* __restrict__ pool, float* __restrict__ cnt) {
    int idx = blockIdx.x * blockDim.x + threadIdx.x;
    if (idx < NGRAPHS * 64) pool[idx] = 0.0f;
    if (idx < NGRAPHS) cnt[idx] = 0.0f;
}

__global__ void k_pool_acc(const float* __restrict__ x4, const int* __restrict__ batch,
                           float* __restrict__ pool, float* __restrict__ cnt) {
    int idx = blockIdx.x * blockDim.x + threadIdx.x;
    if (idx >= NC) return;
    int i = idx >> 6, c = idx & 63;
    int g = batch[i];
    atomicAdd(&pool[g * 64 + c], x4[idx]);
    if (c == 0) atomicAdd(&cnt[g], 1.0f);
}

// ---------------- MLP head + log_softmax ----------------
__global__ void k_head(const float* __restrict__ pool, const float* __restrict__ cnt,
                       const float* __restrict__ lw1, const float* __restrict__ lb1,
                       const float* __restrict__ lw2, const float* __restrict__ lb2,
                       float* __restrict__ out) {
    int g = blockIdx.x * blockDim.x + threadIdx.x;
    if (g >= NGRAPHS) return;
    float inv = 1.0f / fmaxf(cnt[g], 1.0f);
    float l0 = lb2[0], l1 = lb2[1];
    for (int j = 0; j < 32; ++j) {
        float a = lb1[j];
        for (int c = 0; c < 64; ++c)
            a += pool[g * 64 + c] * inv * lw1[c * 32 + j];
        a = fmaxf(a, 0.0f);
        l0 += a * lw2[j * 2 + 0];
        l1 += a * lw2[j * 2 + 1];
    }
    float mx  = fmaxf(l0, l1);
    float lse = mx + logf(expf(l0 - mx) + expf(l1 - mx));
    out[g * 2 + 0] = l0 - lse;
    out[g * 2 + 1] = l1 - lse;
}

extern "C" void kernel_launch(void* const* d_in, const int* in_sizes, int n_in,
                              void* d_out, int out_size, void* d_ws, size_t ws_size,
                              hipStream_t stream) {
    const float* x   = (const float*)d_in[0];
    const float* W1  = (const float*)d_in[1];
    const float* b1  = (const float*)d_in[2];
    const float* W2  = (const float*)d_in[3];
    const float* b2  = (const float*)d_in[4];
    const float* W3  = (const float*)d_in[5];
    const float* b3  = (const float*)d_in[6];
    const float* W4  = (const float*)d_in[7];
    const float* b4  = (const float*)d_in[8];
    const float* g1  = (const float*)d_in[9];
    const float* be1 = (const float*)d_in[10];
    const float* m1  = (const float*)d_in[11];
    const float* v1  = (const float*)d_in[12];
    const float* g2  = (const float*)d_in[13];
    const float* be2 = (const float*)d_in[14];
    const float* m2  = (const float*)d_in[15];
    const float* v2  = (const float*)d_in[16];
    const float* g3  = (const float*)d_in[17];
    const float* be3 = (const float*)d_in[18];
    const float* m3  = (const float*)d_in[19];
    const float* v3  = (const float*)d_in[20];
    const float* lw1 = (const float*)d_in[21];
    const float* lb1 = (const float*)d_in[22];
    const float* lw2 = (const float*)d_in[23];
    const float* lb2 = (const float*)d_in[24];
    const int* src   = (const int*)d_in[25];
    const int* dst   = (const int*)d_in[26];
    const int* batch = (const int*)d_in[27];
    float* out = (float*)d_out;

    const size_t N64 = (size_t)NNODES * 64;
    float* ws      = (float*)d_ws;
    float* f_deg   = ws;                        // N
    float* f_isq   = ws + NNODES;               // N
    float* f_ht    = ws + 2 * (size_t)NNODES;   // N*64 (GEMM output)
    float* f_agg   = f_ht + N64;                // N*64 (aggregation)
    float* f_xA    = f_agg + N64;               // N*64 (activations ping)
    float* f_xB    = f_xA + N64;                // N*64 (activations pong)
    _Float16* f_wf = (_Float16*)(f_xB + N64);   // 3*4096 halfs (W2/W3/W4 fragments)
    float* f_pool  = (float*)(f_wf + 3 * 4096); // G*64
    float* f_cnt   = f_pool + (size_t)NGRAPHS * 64;

    const int bNC = (NC + 255) / 256;
    const int bN  = (NNODES + 255) / 256;
    const int bE  = (NEDGES + 255) / 256;
    const int bSc = (NEDGES * 16 + 255) / 256;  // 16 lanes per edge
    const int bGemm = (NSLABS + 3) / 4;         // 4 waves / 128-thread block

    // graph structure
    k_init_deg<<<bN, 256, 0, stream>>>(f_deg);
    k_count_deg<<<bE, 256, 0, stream>>>(dst, f_deg);
    k_inv_sqrt<<<bN, 256, 0, stream>>>(f_deg, f_isq);
    k_prep_wfrag<<<(3 * 4096 + 255) / 256, 256, 0, stream>>>(W2, W3, W4, f_wf);

    // layer 1
    k_lin1<<<bNC, 256, 0, stream>>>(x, W1, f_isq, b1, f_ht, f_agg);
    k_scatter<<<bSc, 256, 0, stream>>>(f_ht, src, dst, f_isq, f_agg);
    k_bnrelu<<<bNC, 256, 0, stream>>>(f_agg, g1, be1, m1, v1, nullptr, f_xA);

    // layer 2 (+ residual x1)
    k_gemm64_wmma<<<bGemm, 128, 0, stream>>>(f_xA, f_wf + 0 * 4096, f_isq, b2, f_ht, f_agg);
    k_scatter<<<bSc, 256, 0, stream>>>(f_ht, src, dst, f_isq, f_agg);
    k_bnrelu<<<bNC, 256, 0, stream>>>(f_agg, g2, be2, m2, v2, f_xA, f_xB);

    // layer 3 (+ residual x2)
    k_gemm64_wmma<<<bGemm, 128, 0, stream>>>(f_xB, f_wf + 1 * 4096, f_isq, b3, f_ht, f_agg);
    k_scatter<<<bSc, 256, 0, stream>>>(f_ht, src, dst, f_isq, f_agg);
    k_bnrelu<<<bNC, 256, 0, stream>>>(f_agg, g3, be3, m3, v3, f_xB, f_xA);

    // layer 4 (no BN/ReLU): x4 = f_agg
    k_gemm64_wmma<<<bGemm, 128, 0, stream>>>(f_xA, f_wf + 2 * 4096, f_isq, b4, f_ht, f_agg);
    k_scatter<<<bSc, 256, 0, stream>>>(f_ht, src, dst, f_isq, f_agg);

    // pooling + head
    k_pool_zero<<<(NGRAPHS * 64 + 255) / 256, 256, 0, stream>>>(f_pool, f_cnt);
    k_pool_acc<<<bNC, 256, 0, stream>>>(f_agg, batch, f_pool, f_cnt);
    k_head<<<(NGRAPHS + 255) / 256, 256, 0, stream>>>(f_pool, f_cnt, lw1, lb1, lw2, lb2, out);
}
